// SiglipAttention_10187662426835
// MI455X (gfx1250) — matrix operand order repaired and checked by reading, not compiled, not run on hardware
//
#include <hip/hip_runtime.h>
#include <hip/hip_bf16.h>
#include <stdint.h>

typedef __attribute__((ext_vector_type(16))) __bf16 v16bf;
typedef __attribute__((ext_vector_type(8)))  __bf16 v8bf;
typedef __attribute__((ext_vector_type(8)))  float  v8f;
typedef int ivec4 __attribute__((vector_size(4 * sizeof(int))));

#define HIDDEN 1152
#define NH 16
#define HD 72
#define BATCH 8
#define SEQ 1024
#define MROWS (BATCH*SEQ)
#define DP 96   /* head_dim padded to multiple of 32 (WMMA bf16 K) */
#define DV 80   /* head_dim padded to multiple of 16 (WMMA N)      */
#define NKSTEP (HIDDEN/32)

// ---- async global->LDS path (gfx1250), guarded so the build can't break ----
#if defined(__has_builtin)
#if __has_builtin(__builtin_amdgcn_global_load_async_to_lds_b128)
#define HAVE_ASYNC_LDS 1
#endif
#endif

static __device__ __forceinline__ void cp16(const __bf16* g, __bf16* l) {
#if defined(HAVE_ASYNC_LDS)
  // global_load_async_to_lds_b128: per-lane 16B, tracked by ASYNCcnt.
  __builtin_amdgcn_global_load_async_to_lds_b128(
      (__attribute__((address_space(1))) ivec4*)(uintptr_t)g,
      (__attribute__((address_space(3))) ivec4*)(uintptr_t)(uint32_t)(uintptr_t)l,
      0, 0);
#else
  *(v8bf*)l = *(const v8bf*)g;
#endif
}
static __device__ __forceinline__ void wait_async() {
#if defined(HAVE_ASYNC_LDS)
#if __has_builtin(__builtin_amdgcn_s_wait_asynccnt)
  __builtin_amdgcn_s_wait_asynccnt(0);
#else
  asm volatile("s_wait_asynccnt 0" ::: "memory");
#endif
#endif
}

static __device__ __forceinline__ v16bf mk16(v8bf lo, v8bf hi) {
  v16bf r;
#pragma unroll
  for (int i = 0; i < 8; i++) { r[i] = lo[i]; r[i + 8] = hi[i]; }
  return r;
}
static __device__ __forceinline__ v8f vzero8() {
  v8f z;
#pragma unroll
  for (int i = 0; i < 8; i++) z[i] = 0.0f;
  return z;
}
static __device__ __forceinline__ v8f wmma_bf16(v16bf a, v16bf b, v8f c) {
  return __builtin_amdgcn_wmma_f32_16x16x32_bf16(false, a, false, b, (short)0, c,
                                                 false, false);
}

// ---------------------------------------------------------------- converts
__global__ __launch_bounds__(256) void cvt_f32_bf16(const float* __restrict__ src,
                                                    __bf16* __restrict__ dst, int n) {
  int i = blockIdx.x * 256 + threadIdx.x;
  if (i < n) dst[i] = (__bf16)src[i];
}

__global__ __launch_bounds__(256) void cvt_transpose_w(const float* __restrict__ W,
                                                       __bf16* __restrict__ Wt) {
  int i = blockIdx.x * 256 + threadIdx.x;
  int k = i / HIDDEN, n = i - k * HIDDEN;
  Wt[(size_t)n * HIDDEN + k] = (__bf16)W[i];
}

__global__ __launch_bounds__(256) void zero_bf16(__bf16* __restrict__ p, long n) {
  long i = (long)blockIdx.x * 256 + threadIdx.x;
  if (i < n) p[i] = (__bf16)0.0f;
}

// ---------------------------------------------------------------- GEMM
// C[M,N] = A[M,K]*Bt[N,K]^T + bias.  128x128 block tile, BK=32, async
// double-buffered LDS, one barrier per K-step, 8 WMMA per wave per step.
// mode 0: Qp bf16 (x 1/sqrt(72)) | 1: Kp bf16 | 2: Vt bf16 | 3: f32 out
__global__ __launch_bounds__(256) void gemm_bf16_wmma(
    const __bf16* __restrict__ A, const __bf16* __restrict__ Bt,
    const float* __restrict__ bias, void* __restrict__ outp, int mode) {
  __shared__ __align__(16) __bf16 sA[2][128 * 40];
  __shared__ __align__(16) __bf16 sB[2][128 * 40];
  const int tid = threadIdx.x;
  const int wv = tid >> 5, lane = tid & 31, half = lane >> 4, r = lane & 15;
  const int m0 = blockIdx.y * 128, n0 = blockIdx.x * 128;
  const int wm = (wv & 3) * 32;   // wave: 2 M-tiles (32 rows)
  const int wn = (wv >> 2) * 64;  // wave: 4 N-tiles (64 cols)

  v8f acc[2][4];
#pragma unroll
  for (int mt = 0; mt < 2; mt++)
#pragma unroll
    for (int nt = 0; nt < 4; nt++) acc[mt][nt] = vzero8();

  const int rb = tid >> 2, cs = (tid & 3) * 8;  // per-thread copy slot
  auto copy_tile = [&](int k0, int buf) {
    cp16(&A[(size_t)(m0 + rb) * HIDDEN + k0 + cs],       &sA[buf][rb * 40 + cs]);
    cp16(&A[(size_t)(m0 + rb + 64) * HIDDEN + k0 + cs],  &sA[buf][(rb + 64) * 40 + cs]);
    cp16(&Bt[(size_t)(n0 + rb) * HIDDEN + k0 + cs],      &sB[buf][rb * 40 + cs]);
    cp16(&Bt[(size_t)(n0 + rb + 64) * HIDDEN + k0 + cs], &sB[buf][(rb + 64) * 40 + cs]);
  };

  copy_tile(0, 0);
  for (int kk = 0; kk < NKSTEP; kk++) {
    const int cur = kk & 1;
    wait_async();       // own async copies for tile kk complete
    __syncthreads();    // everyone's copies complete; also fences tile kk-1 reads
    if (kk + 1 < NKSTEP) copy_tile((kk + 1) * 32, cur ^ 1);

    v16bf a[2], b[4];
#pragma unroll
    for (int mt = 0; mt < 2; mt++) {
      const __bf16* pa = &sA[cur][(wm + mt * 16 + r) * 40 + half * 8];
      a[mt] = mk16(*(const v8bf*)pa, *(const v8bf*)(pa + 16));
    }
#pragma unroll
    for (int nt = 0; nt < 4; nt++) {
      const __bf16* pb = &sB[cur][(wn + nt * 16 + r) * 40 + half * 8];
      b[nt] = mk16(*(const v8bf*)pb, *(const v8bf*)(pb + 16));
    }
#pragma unroll
    for (int mt = 0; mt < 2; mt++)
#pragma unroll
      for (int nt = 0; nt < 4; nt++)
        acc[mt][nt] = wmma_bf16(a[mt], b[nt], acc[mt][nt]);
  }

#pragma unroll
  for (int mt = 0; mt < 2; mt++) {
#pragma unroll
    for (int nt = 0; nt < 4; nt++) {
#pragma unroll
      for (int rr = 0; rr < 8; rr++) {
        int m = m0 + wm + mt * 16 + rr + 8 * half;  // C/D: row = r + 8*half
        int n = n0 + wn + nt * 16 + r;              // col = lane&15
        float v = acc[mt][nt][rr] + bias[n];
        if (mode == 3) {
          ((float*)outp)[(size_t)m * HIDDEN + n] = v;
        } else {
          int b_ = m >> 10, s = m & 1023;
          int h = n / HD, d = n - h * HD;
          if (mode == 2) {
            ((__bf16*)outp)[((size_t)(b_ * NH + h) * DV + d) * SEQ + s] = (__bf16)v;
          } else {
            if (mode == 0) v *= 0.11785113019775793f;  // 1/sqrt(72)
            ((__bf16*)outp)[((size_t)(b_ * NH + h) * SEQ + s) * DP + d] = (__bf16)v;
          }
        }
      }
    }
  }
}

// ---------------------------------------------------------------- attention
__global__ __launch_bounds__(256) void attn_fa(
    const __bf16* __restrict__ Qp, const __bf16* __restrict__ Kp,
    const __bf16* __restrict__ Vt, __bf16* __restrict__ AOb) {
  __shared__ __align__(16) __bf16 sP[8][16 * 40];
  const int tid = threadIdx.x;
  const int wv = tid >> 5, lane = tid & 31, half = lane >> 4, r = lane & 15;
  const int bh = blockIdx.y;
  const int qrow0 = blockIdx.x * 128 + wv * 16;
  const __bf16* Qb = Qp + (size_t)bh * SEQ * DP;
  const __bf16* Kb = Kp + (size_t)bh * SEQ * DP;
  const __bf16* Vb = Vt + (size_t)bh * DV * SEQ;

  v16bf qf[3];
  {
    int qs = qrow0 + r;
#pragma unroll
    for (int c = 0; c < 3; c++) {
      const __bf16* p = &Qb[(size_t)qs * DP + c * 32 + half * 8];
      qf[c] = mk16(*(const v8bf*)p, *(const v8bf*)(p + 16));
    }
  }

  float m_[8], l_[8];
  v8f O[5];
#pragma unroll
  for (int i = 0; i < 8; i++) { m_[i] = -1e30f; l_[i] = 0.0f; }
#pragma unroll
  for (int t = 0; t < 5; t++) O[t] = vzero8();

  for (int j0 = 0; j0 < SEQ; j0 += 32) {
    // batch-load all 6 K fragments, then 6 back-to-back WMMAs
    v16bf kf[2][3];
#pragma unroll
    for (int nt = 0; nt < 2; nt++) {
      int ks = j0 + nt * 16 + r;
#pragma unroll
      for (int c = 0; c < 3; c++) {
        const __bf16* p = &Kb[(size_t)ks * DP + c * 32 + half * 8];
        kf[nt][c] = mk16(*(const v8bf*)p, *(const v8bf*)(p + 16));
      }
    }
    v8f st[2];
#pragma unroll
    for (int nt = 0; nt < 2; nt++) {
      v8f a = vzero8();
#pragma unroll
      for (int c = 0; c < 3; c++) a = wmma_bf16(qf[c], kf[nt][c], a);
      st[nt] = a;
    }

    // online softmax (rows live in 16-lane half-groups)
    float alpha[8];
#pragma unroll
    for (int rr = 0; rr < 8; rr++) {
      float mx = fmaxf(st[0][rr], st[1][rr]);
      mx = fmaxf(mx, __shfl_xor(mx, 1));
      mx = fmaxf(mx, __shfl_xor(mx, 2));
      mx = fmaxf(mx, __shfl_xor(mx, 4));
      mx = fmaxf(mx, __shfl_xor(mx, 8));
      float mn = fmaxf(m_[rr], mx);
      alpha[rr] = __expf(m_[rr] - mn);
      m_[rr] = mn;
      float p0 = __expf(st[0][rr] - mn);
      float p1 = __expf(st[1][rr] - mn);
      st[0][rr] = p0; st[1][rr] = p1;
      float rs = p0 + p1;
      rs += __shfl_xor(rs, 1);
      rs += __shfl_xor(rs, 2);
      rs += __shfl_xor(rs, 4);
      rs += __shfl_xor(rs, 8);
      l_[rr] = l_[rr] * alpha[rr] + rs;
    }
#pragma unroll
    for (int t = 0; t < 5; t++)
#pragma unroll
      for (int rr = 0; rr < 8; rr++) O[t][rr] *= alpha[rr];

    // redistribute P: C/D layout -> A layout via private LDS slice
#pragma unroll
    for (int rr = 0; rr < 8; rr++) {
      int row = rr + 8 * half;
      sP[wv][row * 40 + r]      = (__bf16)st[0][rr];
      sP[wv][row * 40 + 16 + r] = (__bf16)st[1][rr];
    }
    __builtin_amdgcn_wave_barrier();  // DS ops are in-order per wave
    const __bf16* pp = &sP[wv][r * 40 + half * 8];
    v16bf pf = mk16(*(const v8bf*)pp, *(const v8bf*)(pp + 16));
    __builtin_amdgcn_wave_barrier();

    // batch-load 5 V fragments, then 5 WMMAs
    v16bf vf[5];
#pragma unroll
    for (int t = 0; t < 5; t++) {
      const __bf16* vp = &Vb[(size_t)(t * 16 + r) * SEQ + j0 + half * 8];
      vf[t] = mk16(*(const v8bf*)vp, *(const v8bf*)(vp + 16));
    }
#pragma unroll
    for (int t = 0; t < 5; t++) O[t] = wmma_bf16(pf, vf[t], O[t]);
  }

  int b = bh >> 4, h = bh & 15;
#pragma unroll
  for (int t = 0; t < 5; t++) {
#pragma unroll
    for (int rr = 0; rr < 8; rr++) {
      int d = t * 16 + r;
      if (d < HD) {
        int s = qrow0 + rr + 8 * half;
        float v = O[t][rr] / l_[rr];
        AOb[(size_t)(b * SEQ + s) * HIDDEN + h * HD + d] = (__bf16)v;
      }
    }
  }
}

// ---------------------------------------------------------------- launch
extern "C" void kernel_launch(void* const* d_in, const int* in_sizes, int n_in,
                              void* d_out, int out_size, void* d_ws, size_t ws_size,
                              hipStream_t stream) {
  const float* x  = (const float*)d_in[0];
  const float* wq = (const float*)d_in[1];
  const float* bq = (const float*)d_in[2];
  const float* wk = (const float*)d_in[3];
  const float* bk = (const float*)d_in[4];
  const float* wvp = (const float*)d_in[5];
  const float* bv = (const float*)d_in[6];
  const float* wo = (const float*)d_in[7];
  const float* bo = (const float*)d_in[8];

  char* w = (char*)d_ws;
  size_t off = 0;
  auto take = [&](size_t bytes) -> char* {
    char* p = w + off;
    off += (bytes + 255) & ~(size_t)255;
    return p;
  };
  __bf16* xb  = (__bf16*)take((size_t)MROWS * HIDDEN * 2);
  __bf16* wtq = (__bf16*)take((size_t)HIDDEN * HIDDEN * 2);
  __bf16* wtk = (__bf16*)take((size_t)HIDDEN * HIDDEN * 2);
  __bf16* wtv = (__bf16*)take((size_t)HIDDEN * HIDDEN * 2);
  __bf16* wto = (__bf16*)take((size_t)HIDDEN * HIDDEN * 2);
  __bf16* Qp  = (__bf16*)take((size_t)BATCH * NH * SEQ * DP * 2);
  __bf16* Kp  = (__bf16*)take((size_t)BATCH * NH * SEQ * DP * 2);
  __bf16* Vt  = (__bf16*)take((size_t)BATCH * NH * DV * SEQ * 2);
  __bf16* AOb = (__bf16*)take((size_t)MROWS * HIDDEN * 2);

  cvt_f32_bf16<<<(MROWS * HIDDEN) / 256, 256, 0, stream>>>(x, xb, MROWS * HIDDEN);
  cvt_transpose_w<<<(HIDDEN * HIDDEN) / 256, 256, 0, stream>>>(wq, wtq);
  cvt_transpose_w<<<(HIDDEN * HIDDEN) / 256, 256, 0, stream>>>(wk, wtk);
  cvt_transpose_w<<<(HIDDEN * HIDDEN) / 256, 256, 0, stream>>>(wvp, wtv);
  cvt_transpose_w<<<(HIDDEN * HIDDEN) / 256, 256, 0, stream>>>(wo, wto);

  long nQK = (long)BATCH * NH * SEQ * DP;
  long nV  = (long)BATCH * NH * DV * SEQ;
  zero_bf16<<<(unsigned)((nQK + 255) / 256), 256, 0, stream>>>(Qp, nQK);
  zero_bf16<<<(unsigned)((nQK + 255) / 256), 256, 0, stream>>>(Kp, nQK);
  zero_bf16<<<(unsigned)((nV + 255) / 256), 256, 0, stream>>>(Vt, nV);

  dim3 gg(HIDDEN / 128, MROWS / 128);
  gemm_bf16_wmma<<<gg, 256, 0, stream>>>(xb, wtq, bq, Qp, 0);
  gemm_bf16_wmma<<<gg, 256, 0, stream>>>(xb, wtk, bk, Kp, 1);
  gemm_bf16_wmma<<<gg, 256, 0, stream>>>(xb, wtv, bv, Vt, 2);

  attn_fa<<<dim3(SEQ / 128, BATCH * NH), 256, 0, stream>>>(Qp, Kp, Vt, AOb);

  gemm_bf16_wmma<<<gg, 256, 0, stream>>>(AOb, wto, bo, d_out, 3);
}
